// PEM_67757404061751
// MI455X (gfx1250) — matrix operand order repaired and checked by reading, not compiled, not run on hardware
//
#include <hip/hip_runtime.h>
#include <hip/hip_bf16.h>
#include <math.h>

// ---------------------------------------------------------------------------
// Types for WMMA fragments (gfx1250 wave32).
// ---------------------------------------------------------------------------
typedef __attribute__((ext_vector_type(16))) _Float16 v16h;
typedef __attribute__((ext_vector_type(8)))  _Float16 v8h;
typedef __attribute__((ext_vector_type(4)))  _Float16 v4h;
typedef __attribute__((ext_vector_type(8)))  float    v8f;

#define DIM   1024
#define N1    32768
#define N2    512
#define TOPK  64

// ---------------------------------------------------------------------------
// Kernel 1: fused LayerNorm + cosine L2-normalize + f32->f16 downconvert.
// One 256-thread block per row; each thread owns 4 contiguous elements.
// Single pass over the 128 MB input (the HBM-bound part of the problem).
// ---------------------------------------------------------------------------
__global__ __launch_bounds__(256) void ln_l2_f16_kernel(
    const float* __restrict__ in, const float* __restrict__ gamma,
    const float* __restrict__ beta, _Float16* __restrict__ out)
{
  __shared__ float red[256];
  const int row = blockIdx.x;
  const int t   = threadIdx.x;
  const float4 x = *(const float4*)(in + (size_t)row * DIM + t * 4);

  // mean
  red[t] = x.x + x.y + x.z + x.w;
  __syncthreads();
  for (int s = 128; s > 0; s >>= 1) { if (t < s) red[t] += red[t + s]; __syncthreads(); }
  const float mu = red[0] * (1.0f / DIM);
  __syncthreads();

  // variance
  const float dx = x.x - mu, dy = x.y - mu, dz = x.z - mu, dw = x.w - mu;
  red[t] = dx*dx + dy*dy + dz*dz + dw*dw;
  __syncthreads();
  for (int s = 128; s > 0; s >>= 1) { if (t < s) red[t] += red[t + s]; __syncthreads(); }
  const float rstd = rsqrtf(red[0] * (1.0f / DIM) + 1e-5f);
  __syncthreads();

  // LN affine
  const float4 g = *(const float4*)(gamma + t * 4);
  const float4 b = *(const float4*)(beta  + t * 4);
  const float y0 = dx * rstd * g.x + b.x;
  const float y1 = dy * rstd * g.y + b.y;
  const float y2 = dz * rstd * g.z + b.z;
  const float y3 = dw * rstd * g.w + b.w;

  // L2 norm of the LN output (cosine_similarity denominator)
  red[t] = y0*y0 + y1*y1 + y2*y2 + y3*y3;
  __syncthreads();
  for (int s = 128; s > 0; s >>= 1) { if (t < s) red[t] += red[t + s]; __syncthreads(); }
  const float scale = 1.0f / fmaxf(sqrtf(red[0]), 1e-8f);

  v4h o;
  o[0] = (_Float16)(y0 * scale);
  o[1] = (_Float16)(y1 * scale);
  o[2] = (_Float16)(y2 * scale);
  o[3] = (_Float16)(y3 * scale);
  *(v4h*)(out + (size_t)row * DIM + t * 4) = o;
}

// ---------------------------------------------------------------------------
// Kernel 2: sim[512,32768] = A[512,1024] * B[32768,1024]^T  (f16 in, f32 out)
// via v_wmma_f32_16x16x32_f16.
//
// Each wave computes a 32(M) x 64(N) block: 2 A fragments x 4 B fragments
// -> 8 WMMAs per 6 fragment loads (24 B/lane/WMMA instead of 64 B), since
// the GEMM is L2-bandwidth bound (x1n/f16 = 64 MB lives in the 192 MB L2).
// The 8 waves of a block share one N-block so B lines are reused in L0.
//
// Fragment layouts as loaded here (wave32; A and B both in contiguous k order):
//   A 16x32: lane L holds row M=L%16; lanes 0-15 K=k0..k0+15,
//            lanes 16-31 K=k0+16..k0+31 -> two contiguous 16B loads.
//   B 32x16: lane L holds col N=L%16; lanes 0-15 K=k0..k0+15, lanes 16-31
//            K=k0+16..k0+31 -> B column n is x1 row n: 16 contiguous f16.
//   D: VGPR r -> M = r + 8*(lane>=16), N = lane%16.
// ---------------------------------------------------------------------------
__global__ __launch_bounds__(256) void wmma_sim_kernel(
    const _Float16* __restrict__ A,   // x2n  [N2, DIM]
    const _Float16* __restrict__ B,   // x1n  [N1, DIM]
    float* __restrict__ sim)          // [N2, N1]
{
  const int wave = threadIdx.x >> 5;
  const int lane = threadIdx.x & 31;
  const int tile = blockIdx.x * 8 + wave;   // 8192 wave-blocks
  const int mblk = tile & 15;               // 16 M-blocks of 32 rows
  const int nblk = tile >> 4;               // 512 N-blocks of 64 cols
  const int half = lane >> 4;               // 0: lanes 0-15, 1: lanes 16-31
  const int l15  = lane & 15;

  const _Float16* arow = A + (size_t)(mblk * 32 + l15) * DIM + half * 16;
  const _Float16* brow = B + (size_t)(nblk * 64 + l15) * DIM + half * 16;

  v8f c[2][4] = {};

  #pragma unroll 2
  for (int k0 = 0; k0 < DIM; k0 += 32) {
    v16h af[2];
    #pragma unroll
    for (int i = 0; i < 2; ++i) {
      const v8h alo = *(const v8h*)(arow + (size_t)i * 16 * DIM + k0);
      const v8h ahi = *(const v8h*)(arow + (size_t)i * 16 * DIM + k0 + 8);
      af[i] = __builtin_shufflevector(alo, ahi,
          0, 1, 2, 3, 4, 5, 6, 7, 8, 9, 10, 11, 12, 13, 14, 15);
    }
    v16h bf[4];
    #pragma unroll
    for (int j = 0; j < 4; ++j)
      bf[j] = *(const v16h*)(brow + (size_t)j * 16 * DIM + k0);

    #pragma unroll
    for (int i = 0; i < 2; ++i)
      #pragma unroll
      for (int j = 0; j < 4; ++j)
        c[i][j] = __builtin_amdgcn_wmma_f32_16x16x32_f16(
            /*neg_a=*/false, af[i], /*neg_b=*/false, bf[j],
            /*c_mod=*/(short)0, c[i][j], /*reuse_a=*/false, /*reuse_b=*/false);
  }

  #pragma unroll
  for (int i = 0; i < 2; ++i) {
    #pragma unroll
    for (int j = 0; j < 4; ++j) {
      float* drow = sim + (size_t)(mblk * 32 + i * 16 + half * 8) * N1
                        + nblk * 64 + j * 16 + l15;
      #pragma unroll
      for (int r = 0; r < 8; ++r) drow[(size_t)r * N1] = c[i][j][r];
    }
  }
}

// ---------------------------------------------------------------------------
// Kernel 3: per-row top-64 of 32768 values. One 1024-thread block per row;
// each thread keeps 32 row elements in registers (stride-1024 -> coalesced).
// 64 iterations of block-wide argmax (ties -> smallest index, matching
// jax.lax.top_k / torch.topk), winner zaps its local copy.
// ---------------------------------------------------------------------------
__global__ __launch_bounds__(1024) void topk_kernel(
    const float* __restrict__ sim, float* __restrict__ out)
{
  __shared__ float rv[1024];
  __shared__ int   ri[1024];
  const int row = blockIdx.x;
  const int t   = threadIdx.x;
  const float* srow = sim + (size_t)row * N1;

  float v[32];
  #pragma unroll
  for (int j = 0; j < 32; ++j) v[j] = srow[j * 1024 + t];

  for (int it = 0; it < TOPK; ++it) {
    // local argmax; strict '>' keeps the smallest j (smallest global index)
    float bv = v[0]; int bj = 0;
    #pragma unroll
    for (int j = 1; j < 32; ++j) if (v[j] > bv) { bv = v[j]; bj = j; }

    rv[t] = bv; ri[t] = bj * 1024 + t;
    __syncthreads();
    for (int s = 512; s > 0; s >>= 1) {
      if (t < s) {
        const float ov = rv[t + s]; const int oi = ri[t + s];
        if (ov > rv[t] || (ov == rv[t] && oi < ri[t])) { rv[t] = ov; ri[t] = oi; }
      }
      __syncthreads();
    }

    const int widx = ri[0];
    if (t == 0) out[row * TOPK + it] = (float)widx * (1.0f / (float)N1);

    if ((widx & 1023) == t) {               // owning thread removes the winner
      const int kj = widx >> 10;
      #pragma unroll
      for (int j = 0; j < 32; ++j) if (j == kj) v[j] = -INFINITY;
    }
    __syncthreads();
  }
}

// ---------------------------------------------------------------------------
// Launch: LN(prior)->f16 | LN(clip)->f16 | WMMA sim | top-k.
// Workspace: x1h 64MB | x2h 1MB | sim 64MB  (x1h + sim are L2-resident).
// ---------------------------------------------------------------------------
extern "C" void kernel_launch(void* const* d_in, const int* in_sizes, int n_in,
                              void* d_out, int out_size, void* d_ws, size_t ws_size,
                              hipStream_t stream) {
  const float* prior = (const float*)d_in[0];
  const float* clip  = (const float*)d_in[1];
  const float* g1    = (const float*)d_in[2];
  const float* b1    = (const float*)d_in[3];
  const float* g2    = (const float*)d_in[4];
  const float* b2    = (const float*)d_in[5];
  float* out = (float*)d_out;

  char* ws = (char*)d_ws;
  _Float16* x1h = (_Float16*)ws;                                           // 64 MB
  _Float16* x2h = (_Float16*)(ws + (size_t)N1 * DIM * 2);                  //  1 MB
  float*    sim = (float*)   (ws + (size_t)N1 * DIM * 2 + (size_t)N2 * DIM * 2);

  ln_l2_f16_kernel<<<N1, 256, 0, stream>>>(prior, g1, b1, x1h);
  ln_l2_f16_kernel<<<N2, 256, 0, stream>>>(clip,  g2, b2, x2h);

  // 16 M-blocks (32 rows) * 512 N-blocks (64 cols) = 8192 wave-blocks,
  // 8 waves per 256-thread block -> 1024 blocks.
  wmma_sim_kernel<<<1024, 256, 0, stream>>>(x2h, x1h, sim);

  topk_kernel<<<N2, 1024, 0, stream>>>(sim, out);
}
